// LineCNNSimple_27401891349183
// MI455X (gfx1250) — compile-verified, matrix operand order checked by reading, not run on hardware
//
#include <hip/hip_runtime.h>
#include <hip/hip_bf16.h>
#include <stdint.h>

// ---------------------------------------------------------------------------
// MI455X (gfx1250) fused LineCNN: conv1(VALU) + conv2(WMMA bf16) + pool fused
// per image; fc1/fc2 as batched WMMA GEMMs across the 1024 images.
// All matrix math on v_wmma_f32_16x16x32_bf16 (f32 accumulate).
// conv2 processes 2 N-tiles per wave iteration: 4 A frags + 2 B frags feed
// 8 independent WMMAs, hiding ds latency instead of wait-per-wmma.
// ---------------------------------------------------------------------------

typedef __bf16 bf16_t;
typedef bf16_t bf16x8 __attribute__((ext_vector_type(8)));
typedef bf16_t v16bf  __attribute__((ext_vector_type(16)));
typedef float  v8f    __attribute__((ext_vector_type(8)));

#define NIMG  1024     // B*S = 32*32 independent 28x28 images
#define CD    64       // conv channels
#define HH    28
#define WWIN  28
#define PH    30       // padded
#define PW    30
#define NPIX  784      // 28*28
#define Y2STR 800      // y2 row stride: 784 real pixels + 16 slack columns
#define FCIN  12544    // 64*14*14
#define FCD   128
#define NCLS  83
#define NCLSP 96       // 83 padded to 6 M-tiles of 16
#define KC1   392      // 12544 / 32

// ---- fragment helpers ------------------------------------------------------
// 16-bit WMMA operand K-layout (ISA 7.12.2): lanes 0-15 hold K{0..7} then
// K{16..23}; lanes 16-31 hold K{8..15} then K{24..31}. With K contiguous in
// memory that is two 16-byte loads at +0 and +32 bytes from (base + hi*8).
static __device__ __forceinline__ v16bf ld16(const bf16_t* p) {
  union { v16bf v; bf16x8 h[2]; } u;
  u.h[0] = *reinterpret_cast<const bf16x8*>(p);
  u.h[1] = *reinterpret_cast<const bf16x8*>(p + 16);
  return u.v;
}
static __device__ __forceinline__ v8f wmma_bf16(v16bf a, v16bf b, v8f c) {
  return __builtin_amdgcn_wmma_f32_16x16x32_bf16(false, a, false, b,
                                                 (short)0, c, false, false);
}

// ---- LDS layout for the fused conv kernel (dynamic shared, 295,184 B) ------
#define OFF_XPAD 0          // 900  f32  = 3600 B
#define OFF_Y1   3600       // 30*30*64 bf16 (NHWC, padded) = 115200 B
#define OFF_W2   118800     // 9*64*64  bf16 [tap][oc][ic]  = 73728 B
#define OFF_B2   192528     // 64 f32                        = 256 B
#define OFF_Y2   192784     // 64*800  bf16 [oc][pix+slack]  = 102400 B
#define SMEM_BYTES 295184   // < 320 KB WGP LDS

__global__ void __launch_bounds__(256)
conv_fused_kernel(const float* __restrict__ x,
                  const float* __restrict__ w1, const float* __restrict__ b1,
                  const float* __restrict__ w2, const float* __restrict__ b2,
                  bf16_t* __restrict__ feat) {
  extern __shared__ char smem[];
  float*  xpad = (float*)(smem + OFF_XPAD);
  bf16_t* y1   = (bf16_t*)(smem + OFF_Y1);
  bf16_t* w2s  = (bf16_t*)(smem + OFF_W2);
  float*  b2s  = (float*)(smem + OFF_B2);
  bf16_t* y2   = (bf16_t*)(smem + OFF_Y2);

  const int n    = blockIdx.x;       // image id = b*32 + s
  const int bimg = n >> 5, s = n & 31;
  const int tid  = threadIdx.x;

  // ---- stage input window, zero-padded 30x30 fp32 ----
  for (int i = tid; i < PH * PW; i += 256) {
    int ph = i / PW, pw = i - ph * PW;
    int h = ph - 1, w = pw - 1;
    float v = 0.f;
    if ((unsigned)h < HH && (unsigned)w < WWIN)
      v = x[bimg * (HH * 896) + h * 896 + s * WWIN + w];
    xpad[i] = v;
  }
  // ---- zero y1 (covers halo) ----
  {
    uint32_t* y1w = (uint32_t*)y1;
    for (int i = tid; i < (PH * PW * CD) / 2; i += 256) y1w[i] = 0u;
  }
  // ---- stage conv2 weights fp32->bf16, relaid as [tap][oc][ic] ----
  for (int i = tid; i < CD * CD * 9; i += 256) {
    int oc = i / (CD * 9);
    int r  = i - oc * (CD * 9);
    int ic = r / 9;
    int off = r - ic * 9;                       // off = ky*3+kx
    w2s[(off * CD + oc) * CD + ic] = (bf16_t)w2[i];
  }
  if (tid < CD) b2s[tid] = b2[tid];
  __syncthreads();

  // ---- conv1 (C=1, K=9: too small for WMMA) in VALU fp32 + ReLU -> bf16 ----
  {
    const int oc = tid >> 2, q = tid & 3;       // 4 threads share one oc
    float wl[9];
#pragma unroll
    for (int k = 0; k < 9; ++k) wl[k] = w1[oc * 9 + k];
    const float bias = b1[oc];
    for (int p = q; p < NPIX; p += 4) {
      int h = p / WWIN, w = p - h * WWIN;
      float acc = bias;
#pragma unroll
      for (int ky = 0; ky < 3; ++ky)
#pragma unroll
        for (int kx = 0; kx < 3; ++kx)
          acc = fmaf(wl[ky * 3 + kx], xpad[(h + ky) * PW + (w + kx)], acc);
      acc = fmaxf(acc, 0.f);
      y1[((h + 1) * PW + (w + 1)) * CD + oc] = (bf16_t)acc;   // NHWC padded
    }
  }
  __syncthreads();

  // ---- conv2 implicit GEMM: M=64(oc) x K=576(ic*9) x N=784(pix), WMMA ----
  // Two 16-pixel N-tiles per iteration: 4 A frags + 2 B frags -> 8 WMMAs.
  // Pair 24 covers pixels 768..799; 784..799 are slack (y2 has 800-col
  // stride; the corresponding y1 reads land in w2s, still inside LDS).
  {
    const int wave = tid >> 5, lane = tid & 31;
    const int ncol = lane & 15, hi = lane >> 4;
    for (int ntp = wave; ntp < 25; ntp += 8) {       // 25 pairs of N-tiles
      const int px0 = ntp * 32 + ncol;
      const int px1 = px0 + 16;
      const int h0 = px0 / WWIN, w0 = px0 - h0 * WWIN;
      const int h1 = px1 / WWIN, w1x = px1 - h1 * WWIN;
      v8f acc[8] = {};                                // [mt] and [4+mt]
#pragma unroll
      for (int off = 0; off < 9; ++off) {
        const int dy = off / 3, dx = off - dy * 3;
        const bf16_t* brow0 = &y1[((h0 + dy) * PW + (w0 + dx)) * CD];
        const bf16_t* brow1 = &y1[((h1 + dy) * PW + (w1x + dx)) * CD];
        const bf16_t* abase = &w2s[off * CD * CD];
#pragma unroll
        for (int kc = 0; kc < 2; ++kc) {              // 64 ic = 2 K-chunks
          const int kb = kc * 32 + hi * 8;
          v16bf a0 = ld16(abase + (0 * 16 + ncol) * CD + kb);
          v16bf a1 = ld16(abase + (1 * 16 + ncol) * CD + kb);
          v16bf a2 = ld16(abase + (2 * 16 + ncol) * CD + kb);
          v16bf a3 = ld16(abase + (3 * 16 + ncol) * CD + kb);
          v16bf bf0 = ld16(brow0 + kb);
          v16bf bf1 = ld16(brow1 + kb);
          acc[0] = wmma_bf16(a0, bf0, acc[0]);
          acc[1] = wmma_bf16(a1, bf0, acc[1]);
          acc[2] = wmma_bf16(a2, bf0, acc[2]);
          acc[3] = wmma_bf16(a3, bf0, acc[3]);
          acc[4] = wmma_bf16(a0, bf1, acc[4]);
          acc[5] = wmma_bf16(a1, bf1, acc[5]);
          acc[6] = wmma_bf16(a2, bf1, acc[6]);
          acc[7] = wmma_bf16(a3, bf1, acc[7]);
        }
      }
      // bias + ReLU, store y2[oc][pix] bf16.  C layout: VGPR r -> M=r (+8 hi)
#pragma unroll
      for (int mt = 0; mt < 4; ++mt)
#pragma unroll
        for (int r = 0; r < 8; ++r) {
          const int oc = mt * 16 + hi * 8 + r;
          const float bb = b2s[oc];
          y2[oc * Y2STR + px0] = (bf16_t)fmaxf(acc[mt][r] + bb, 0.f);
          y2[oc * Y2STR + px1] = (bf16_t)fmaxf(acc[4 + mt][r] + bb, 0.f);
        }
    }
  }
  __syncthreads();

  // ---- 2x2 maxpool, flatten in torch order (ch*196 + py*14 + px) ----
  for (int f = tid; f < FCIN; f += 256) {
    int ch = f / 196, r = f - ch * 196;
    int py = r / 14, pxx = r - py * 14;
    const bf16_t* p0 = &y2[ch * Y2STR + (2 * py) * WWIN + 2 * pxx];
    float m = (float)p0[0];
    m = fmaxf(m, (float)p0[1]);
    m = fmaxf(m, (float)p0[WWIN]);
    m = fmaxf(m, (float)p0[WWIN + 1]);
    feat[n * FCIN + f] = (bf16_t)m;
  }
}

// ---- fp32 -> bf16 weight prep for the FC GEMMs -----------------------------
__global__ void prep_weights_kernel(const float* __restrict__ fc1w,
                                    const float* __restrict__ fc2w,
                                    bf16_t* __restrict__ fc1bf,
                                    bf16_t* __restrict__ fc2bf) {
  const int total1 = FCD * FCIN;
  const int total2 = NCLSP * FCD;
  for (int i = blockIdx.x * blockDim.x + threadIdx.x; i < total1 + total2;
       i += gridDim.x * blockDim.x) {
    if (i < total1) {
      fc1bf[i] = (bf16_t)fc1w[i];
    } else {
      int j = i - total1;
      int row = j / FCD, col = j - row * FCD;
      fc2bf[j] = (row < NCLS) ? (bf16_t)fc2w[row * FCD + col] : (bf16_t)0.f;
    }
  }
}

// ---- fc1: (128 x 12544) @ (12544 x 1024) + bias, ReLU -> bf16 --------------
__global__ void __launch_bounds__(256)
fc1_kernel(const bf16_t* __restrict__ feat, const bf16_t* __restrict__ fc1bf,
           const float* __restrict__ fc1b, bf16_t* __restrict__ hbuf) {
  const int tid = threadIdx.x;
  const int wave = tid >> 5, lane = tid & 31;
  const int ncol = lane & 15, hi = lane >> 4;
  const int mrow = wave * 16 + ncol;              // wave == M-tile (8x16=128)
  const int nbase = blockIdx.x * 32;              // 2 image tiles per block
  const int img0 = nbase + ncol, img1 = nbase + 16 + ncol;

  v8f acc0 = {}, acc1 = {};
#pragma unroll 2
  for (int kc = 0; kc < KC1; ++kc) {
    const int kb = kc * 32 + hi * 8;
    v16bf a  = ld16(&fc1bf[mrow * FCIN + kb]);
    v16bf b0 = ld16(&feat[img0 * FCIN + kb]);
    v16bf b1 = ld16(&feat[img1 * FCIN + kb]);
    acc0 = wmma_bf16(a, b0, acc0);
    acc1 = wmma_bf16(a, b1, acc1);
  }
#pragma unroll
  for (int r = 0; r < 8; ++r) {
    const int fr = wave * 16 + hi * 8 + r;
    const float bb = fc1b[fr];
    hbuf[img0 * FCD + fr] = (bf16_t)fmaxf(acc0[r] + bb, 0.f);
    hbuf[img1 * FCD + fr] = (bf16_t)fmaxf(acc1[r] + bb, 0.f);
  }
}

// ---- fc2: (96pad x 128) @ (128 x 1024) + bias -> out (B, 83, S) fp32 -------
__global__ void __launch_bounds__(256)
fc2_kernel(const bf16_t* __restrict__ hbuf, const bf16_t* __restrict__ fc2bf,
           const float* __restrict__ fc2b, float* __restrict__ out) {
  const int tid = threadIdx.x;
  const int wave = tid >> 5, lane = tid & 31;
  const int ncol = lane & 15, hi = lane >> 4;
  const int img = blockIdx.x * 128 + wave * 16 + ncol;  // wave == N-tile

  v8f acc[6] = {};
#pragma unroll
  for (int kc = 0; kc < 4; ++kc) {                      // K = 128
    const int kb = kc * 32 + hi * 8;
    v16bf b = ld16(&hbuf[img * FCD + kb]);
#pragma unroll
    for (int mt = 0; mt < 6; ++mt) {
      v16bf a = ld16(&fc2bf[(mt * 16 + ncol) * FCD + kb]);
      acc[mt] = wmma_bf16(a, b, acc[mt]);
    }
  }
  const int bb = img >> 5, ss = img & 31;
#pragma unroll
  for (int mt = 0; mt < 6; ++mt)
#pragma unroll
    for (int r = 0; r < 8; ++r) {
      int cls = mt * 16 + hi * 8 + r;
      if (cls < NCLS)
        out[(bb * NCLS + cls) * 32 + ss] = acc[mt][r] + fc2b[cls];
    }
}

// ---------------------------------------------------------------------------
extern "C" void kernel_launch(void* const* d_in, const int* in_sizes, int n_in,
                              void* d_out, int out_size, void* d_ws, size_t ws_size,
                              hipStream_t stream) {
  (void)in_sizes; (void)n_in; (void)out_size; (void)ws_size;
  const float* x    = (const float*)d_in[0];
  const float* w1   = (const float*)d_in[1];
  const float* b1   = (const float*)d_in[2];
  const float* w2   = (const float*)d_in[3];
  const float* b2   = (const float*)d_in[4];
  const float* fc1w = (const float*)d_in[5];
  const float* fc1b = (const float*)d_in[6];
  const float* fc2w = (const float*)d_in[7];
  const float* fc2b = (const float*)d_in[8];
  float* out = (float*)d_out;

  // Workspace layout (bytes), all 256-aligned:
  char* ws = (char*)d_ws;
  bf16_t* feat  = (bf16_t*)(ws);                               // 25,690,112
  bf16_t* fc1bf = (bf16_t*)(ws + 25690112);                    //  3,211,264
  bf16_t* fc2bf = (bf16_t*)(ws + 25690112 + 3211264);          //     24,576
  bf16_t* hbuf  = (bf16_t*)(ws + 25690112 + 3211264 + 24576);  //    262,144

  prep_weights_kernel<<<2048, 256, 0, stream>>>(fc1w, fc2w, fc1bf, fc2bf);
  conv_fused_kernel<<<NIMG, 256, SMEM_BYTES, stream>>>(x, w1, b1, w2, b2, feat);
  fc1_kernel<<<NIMG / 32, 256, 0, stream>>>(feat, fc1bf, fc1b, hbuf);
  fc2_kernel<<<NIMG / 128, 256, 0, stream>>>(hbuf, fc2bf, fc2b, out);
}
